// D2LDotProductAttention_7670811591254
// MI455X (gfx1250) — matrix operand order repaired and checked by reading, not compiled, not run on hardware
//
#include <hip/hip_runtime.h>

// Flash-attention, B=32, S=2048, D=64 fp32, key valid-length masking.
// bf16 WMMA (v_wmma_f32_16x16x32_bf16) for QK^T, PV, and softmax row-sums.
// KB=64 key blocks, double-buffered LDS staging (1 barrier/iter),
// DPP-fused max reductions, vectorized (packed-f32) rescale, masked tail.

#define B_BATCH 32
#define S_LEN   2048
#define D_DIM   64
#define KB      64      // keys per block iteration (4 n-tiles)
#define QROWS   128     // query rows per workgroup (8 waves x 16)
#define NWAVES  8

typedef __attribute__((ext_vector_type(16))) __bf16 v16bf;
typedef __attribute__((ext_vector_type(8)))  __bf16 v8bf;
typedef __attribute__((ext_vector_type(2)))  __bf16 v2bf;
typedef __attribute__((ext_vector_type(8)))  float  v8f;

union Frag { v16bf bf; v8bf h[2]; };

__device__ __forceinline__ float fexp2(float x) {
#if __has_builtin(__builtin_amdgcn_exp2f)
    return __builtin_amdgcn_exp2f(x);     // v_exp_f32 (base-2)
#else
    return exp2f(x);
#endif
}

__device__ __forceinline__ float frcp(float x) {
#if __has_builtin(__builtin_amdgcn_rcpf)
    return __builtin_amdgcn_rcpf(x);      // v_rcp_f32
#else
    return 1.0f / x;
#endif
}

// single-instruction maxes (no canonicalize spam)
__device__ __forceinline__ float fmax_raw(float a, float b) {
    float d;
    asm("v_max_num_f32_e32 %0, %1, %2" : "=v"(d) : "v"(a), "v"(b));
    return d;
}
__device__ __forceinline__ float fmax3_raw(float a, float b, float c) {
    float d;
    asm("v_max3_num_f32 %0, %1, %2, %3" : "=v"(d) : "v"(a), "v"(b), "v"(c));
    return d;
}

// 16-lane row max reduction: 4 DPP-fused max instructions (butterfly)
__device__ __forceinline__ float dpp_rowmax(float v) {
    float r = v;
    asm("v_max_num_f32_dpp %0, %0, %0 quad_perm:[1,0,3,2] row_mask:0xf bank_mask:0xf\n\t"
        "v_max_num_f32_dpp %0, %0, %0 quad_perm:[2,3,0,1] row_mask:0xf bank_mask:0xf\n\t"
        "v_max_num_f32_dpp %0, %0, %0 row_half_mirror row_mask:0xf bank_mask:0xf\n\t"
        "v_max_num_f32_dpp %0, %0, %0 row_mirror row_mask:0xf bank_mask:0xf"
        : "+v"(r));
    return r;
}

// A-fragment address order swaps the middle two 8-slot groups vs. slot order.
__device__ __forceinline__ int group_swap(int a) {
    return ((((a >> 3) ^ (a >> 4)) & 1) ? (a ^ 24) : a);
}

__device__ __forceinline__ v8f bwmma(const Frag& a, const Frag& b, v8f c) {
    return __builtin_amdgcn_wmma_f32_16x16x32_bf16(false, a.bf, false, b.bf,
                                                   (short)0, c, false, false);
}

#define NEGF (-3.0e38f)

// in-flight stage data for one key block (per thread)
struct Stage {
    float k[16];            // K: row r=tid>>2, dims (tid&3)*16 .. +15
    float v0[8], v1[8];     // V: key pair (grp*32+n, +16), dims (tid>>5)*8 .. +7
};

__device__ __forceinline__ Stage load_stage(const float* __restrict__ Kb,
                                            const float* __restrict__ Vb,
                                            int kbase, int tid) {
    Stage s;
    const int r = tid >> 2;
    const int c = (tid & 3) * 16;
    const float* kp = Kb + (size_t)(kbase + r) * D_DIM + c;
    #pragma unroll
    for (int j = 0; j < 16; ++j) s.k[j] = kp[j];

    const int n   = tid & 15;
    const int grp = (tid >> 4) & 1;
    const int cb  = (tid >> 5) * 8;
    const float* vp0 = Vb + (size_t)(kbase + grp * 32 + n) * D_DIM + cb;
    #pragma unroll
    for (int j = 0; j < 8; ++j) { s.v0[j] = vp0[j]; s.v1[j] = vp0[16 * D_DIM + j]; }
    return s;
}

__device__ __forceinline__ void store_stage(const Stage& s, __bf16* sK, __bf16* sVt, int tid) {
    const int r = tid >> 2;
    const int c = (tid & 3) * 16;
    v8bf k0, k1;
    #pragma unroll
    for (int j = 0; j < 8; ++j) { k0[j] = (__bf16)s.k[j]; k1[j] = (__bf16)s.k[8 + j]; }
    *(v8bf*)&sK[r * D_DIM + c]     = k0;
    *(v8bf*)&sK[r * D_DIM + c + 8] = k1;

    const int n   = tid & 15;
    const int grp = (tid >> 4) & 1;
    const int cb  = (tid >> 5) * 8;
    const int aV  = grp * 32 + 2 * n;
    #pragma unroll
    for (int j = 0; j < 8; ++j) {
        v2bf pv;
        pv[0] = (__bf16)s.v0[j];
        pv[1] = (__bf16)s.v1[j];
        *(v2bf*)&sVt[(cb + j) * KB + aV] = pv;
    }
}

template<bool MASK>
__device__ __forceinline__ void compute_block(
    int kbase, int vlen, int laneN, int laneHi,
    const __bf16* sK, const __bf16* sVt, __bf16* sPw,
    const Frag (&aq)[2], const Frag& bone,
    v8f (&acc)[4], v8f& lacc, float (&m_run)[8])
{
    // ---- S' = (Q*scale) K^T : 4 n-tiles x 2 k-steps = 8 wmma ----
    v8f st[4];
    #pragma unroll
    for (int t = 0; t < 4; ++t) st[t] = (v8f)0.0f;
    #pragma unroll
    for (int s = 0; s < 2; ++s) {
        #pragma unroll
        for (int t = 0; t < 4; ++t) {
            Frag bk;
            const int key = t * 16 + laneN;
            bk.h[0] = *(const v8bf*)&sK[key * D_DIM + s * 32 + laneHi * 8];
            bk.h[1] = *(const v8bf*)&sK[key * D_DIM + s * 32 + laneHi * 8 + 16];
            st[t] = bwmma(aq[s], bk, st[t]);
        }
    }

    if (MASK) {
        #pragma unroll
        for (int t = 0; t < 4; ++t) {
            const bool vmt = (kbase + t * 16 + laneN) < vlen;
            #pragma unroll
            for (int i = 0; i < 8; ++i) st[t][i] = vmt ? st[t][i] : NEGF;
        }
    }

    // ---- scalar phase: per-row max (DPP butterfly) + alpha ----
    v8f mnewv, alphav;
    #pragma unroll
    for (int i = 0; i < 8; ++i) {
        float mx = fmax_raw(fmax3_raw(st[0][i], st[1][i], st[2][i]), st[3][i]);
        mx = dpp_rowmax(mx);
        const float mold = m_run[i];
        const float mnew = fmax_raw(mold, mx);
        m_run[i]  = mnew;
        mnewv[i]  = mnew;
        alphav[i] = fexp2(mold - mnew);
    }

    // ---- vector phase: shift scores + rescale accumulators (packed f32) ----
    #pragma unroll
    for (int t = 0; t < 4; ++t) st[t] -= mnewv;
    lacc *= alphav;
    #pragma unroll
    for (int t = 0; t < 4; ++t) acc[t] *= alphav;

    // ---- exp + pack + stage P (pair-packed) ----
    const int aP = 2 * laneN;
    #pragma unroll
    for (int i = 0; i < 8; ++i) {
        const float p0 = fexp2(st[0][i]);
        const float p1 = fexp2(st[1][i]);
        const float p2 = fexp2(st[2][i]);
        const float p3 = fexp2(st[3][i]);
        const int row = laneHi * 8 + i;
        v2bf q01, q23;
        q01[0] = (__bf16)p0; q01[1] = (__bf16)p1;
        q23[0] = (__bf16)p2; q23[1] = (__bf16)p3;
        *(v2bf*)&sPw[row * KB + aP]      = q01;
        *(v2bf*)&sPw[row * KB + 32 + aP] = q23;
    }

    // ---- reload P as two A-fragments (keys 0..31 / 32..63) ----
    Frag ap0, ap1;
    ap0.h[0] = *(const v8bf*)&sPw[laneN * KB + laneHi * 16];
    ap0.h[1] = *(const v8bf*)&sPw[laneN * KB + laneHi * 16 + 8];
    ap1.h[0] = *(const v8bf*)&sPw[laneN * KB + 32 + laneHi * 16];
    ap1.h[1] = *(const v8bf*)&sPw[laneN * KB + 32 + laneHi * 16 + 8];

    // ---- l += P * ones : 2 wmma ----
    lacc = bwmma(ap0, bone, lacc);
    lacc = bwmma(ap1, bone, lacc);

    // ---- O += P V : 4 dim-tiles x 2 k-steps = 8 wmma ----
    #pragma unroll
    for (int t = 0; t < 4; ++t) {
        const int dim = t * 16 + laneN;
        Frag bv0, bv1;
        bv0.h[0] = *(const v8bf*)&sVt[dim * KB + laneHi * 16];
        bv0.h[1] = *(const v8bf*)&sVt[dim * KB + laneHi * 16 + 8];
        bv1.h[0] = *(const v8bf*)&sVt[dim * KB + 32 + laneHi * 16];
        bv1.h[1] = *(const v8bf*)&sVt[dim * KB + 32 + laneHi * 16 + 8];
        acc[t] = bwmma(ap0, bv0, acc[t]);
        acc[t] = bwmma(ap1, bv1, acc[t]);
    }
}

__global__ __launch_bounds__(256) void
fattn_kernel(const float* __restrict__ Q, const float* __restrict__ K,
             const float* __restrict__ V, const int* __restrict__ vlens,
             float* __restrict__ O)
{
    __shared__ __align__(16) __bf16 sK [2][KB * D_DIM];     // 2 x 8 KB
    __shared__ __align__(16) __bf16 sVt[2][D_DIM * KB];     // 2 x 8 KB
    __shared__ __align__(16) __bf16 sP [NWAVES * 16 * KB];  // 16 KB

    const int b    = blockIdx.x;
    const int qt   = blockIdx.y;
    const int tid  = threadIdx.x;
    const int wave = tid >> 5;
    const int lane = tid & 31;
    const int laneN  = lane & 15;
    const int laneHi = lane >> 4;

    const int qbase = qt * QROWS + wave * 16;
    const int vlen  = vlens[b];
    // fold 1/sqrt(64) and log2(e) into Q -> raw v_exp_f32 softmax
    const float QSCALE = 0.125f * 1.44269504088896340736f;

    const float* Kb = K + (size_t)b * S_LEN * D_DIM;
    const float* Vb = V + (size_t)b * S_LEN * D_DIM;

    // ---- Q A-fragments (16x32 bf16 layout), pre-scaled ----
    const float* qrow = Q + ((size_t)b * S_LEN + qbase + laneN) * D_DIM;
    Frag aq[2];
    #pragma unroll
    for (int s = 0; s < 2; ++s) {
        const int koff = s * 32 + laneHi * 8;
        #pragma unroll
        for (int e = 0; e < 8; ++e) {
            aq[s].h[0][e] = (__bf16)(qrow[koff + e]      * QSCALE);
            aq[s].h[1][e] = (__bf16)(qrow[koff + 16 + e] * QSCALE);
        }
    }

    // ---- all-ones B fragment for the row-sum wmma ----
    Frag bone;
    #pragma unroll
    for (int e = 0; e < 8; ++e) { bone.h[0][e] = (__bf16)1.0f; bone.h[1][e] = (__bf16)1.0f; }

    // ---- state ----
    v8f acc[4];
    #pragma unroll
    for (int t = 0; t < 4; ++t) acc[t] = (v8f)0.0f;
    v8f lacc = (v8f)0.0f;
    float m_run[8];
    #pragma unroll
    for (int i = 0; i < 8; ++i) m_run[i] = NEGF;

    __bf16* sPw = &sP[wave * 16 * KB];

    // ---- double-buffered pipeline: one barrier per iteration ----
    const int nkb = (vlen + KB - 1) >> 6;      // >= 1 ; only last block may be partial
    {
        Stage s0 = load_stage(Kb, Vb, 0, tid);
        store_stage(s0, sK[0], sVt[0], tid);
    }
    __syncthreads();

    int cur = 0;
    for (int kb = 0; kb < nkb - 1; ++kb) {
        Stage nx = load_stage(Kb, Vb, (kb + 1) * KB, tid);   // global loads issue early
        compute_block<false>(kb * KB, vlen, laneN, laneHi,
                             sK[cur], sVt[cur], sPw, aq, bone, acc, lacc, m_run);
        store_stage(nx, sK[cur ^ 1], sVt[cur ^ 1], tid);     // other buffer: safe since
        __syncthreads();                                     // last iter's barrier passed
        cur ^= 1;
    }
    // masked body is also correct for a full final block (all lanes pass the compare)
    compute_block<true>((nkb - 1) * KB, vlen, laneN, laneHi,
                        sK[cur], sVt[cur], sPw, aq, bone, acc, lacc, m_run);

    // ---- epilogue: normalize and store fp32 ----
    #pragma unroll
    for (int i = 0; i < 8; ++i) {
        const int row  = qbase + laneHi * 8 + i;
        const float nl = frcp(lacc[i]);
        float* orow = O + ((size_t)b * S_LEN + row) * D_DIM;
        #pragma unroll
        for (int t = 0; t < 4; ++t) {
            orow[t * 16 + laneN] = acc[t][i] * nl;
        }
    }
}

extern "C" void kernel_launch(void* const* d_in, const int* in_sizes, int n_in,
                              void* d_out, int out_size, void* d_ws, size_t ws_size,
                              hipStream_t stream) {
    const float* Q  = (const float*)d_in[0];
    const float* K  = (const float*)d_in[1];
    const float* V  = (const float*)d_in[2];
    const int*   VL = (const int*)d_in[3];
    float*       O  = (float*)d_out;

    dim3 grid(B_BATCH, S_LEN / QROWS);   // 32 x 16
    dim3 block(256);                     // 8 waves (wave32)
    fattn_kernel<<<grid, block, 0, stream>>>(Q, K, V, VL, O);
}
